// LSTMClassifier_27290222199121
// MI455X (gfx1250) — matrix operand order repaired and checked by reading, not compiled, not run on hardware
//
#include <hip/hip_runtime.h>

// ---------------------------------------------------------------------------
// LSTM classifier for MI455X (gfx1250): bf16 WMMA persistent-scan design v6.
// B=64, T=2048, D=320, H=256, C=10.
// 16 WGs x 16 waves; WG q owns hidden slice u=[16q,16q+16) across all 4
// gates. Per step: h-fragment loads pinned at loop top (sched_barrier), in
// flight under the x-part WMMAs (LDS-staged); next step's x staged to LDS in
// the post-GEMM window where hfr/acc are dead. One global barrier per step.
// ---------------------------------------------------------------------------

typedef __attribute__((ext_vector_type(16))) __bf16 v16bf;
typedef __attribute__((ext_vector_type(8)))  float  v8f;
typedef unsigned short ushort_t;

#define NB 64
#define NT 2048
#define ND 320
#define NH 256
#define NC 10
#define NWG 16          // persistent workgroups in scan kernel
#define THR 512         // threads per workgroup (16 waves)

// ---- workspace layout (bytes) ----
constexpr size_t OFF_XPK  = 0;                         // ushort[2048*4*10*512]
constexpr size_t OFF_H1PK = 83886080;                  // ushort[2048*4*8*512]
constexpr size_t OFF_WPK0 = OFF_H1PK + 67108864;       // ushort[64*18*512]
constexpr size_t OFF_WPK1 = OFF_WPK0 + 1179648;        // ushort[64*16*512]
constexpr size_t OFF_B0   = OFF_WPK1 + 1048576;        // float[1024]
constexpr size_t OFF_B1   = OFF_B0 + 4096;             // float[1024]
constexpr size_t OFF_HAPK = OFF_B1 + 4096;             // ushort[4*8*512]
constexpr size_t OFF_H2   = OFF_HAPK + 32768;          // float[64*256]
constexpr size_t OFF_CNT  = OFF_H2 + 65536;            // int

__device__ __forceinline__ ushort_t f2bf(float f) {
    unsigned int u = __float_as_uint(f);
    unsigned int r = u + 0x7FFFu + ((u >> 16) & 1u);   // round-to-nearest-even
    return (ushort_t)(r >> 16);
}

// A-fragment index map (16-bit A 16x32, ISA 7.12.2): (lane,j) -> ko
__device__ __forceinline__ int a_ko(int lane, int j) {
    int half = lane >> 4;
    return (j < 8) ? (half * 8 + j) : (16 + half * 8 + (j - 8));
}

// ---------------------------------------------------------------------------
// Pack x (fp32 [B,T,D]) into bf16 A-fragment layout [T][4][10][32][16]
// ---------------------------------------------------------------------------
__global__ void pack_x_kernel(const float* __restrict__ x, ushort_t* __restrict__ xpk) {
    size_t idx = (size_t)blockIdx.x * blockDim.x + threadIdx.x;
    if (idx >= (size_t)NT * 4 * 10 * 512) return;
    int j    = (int)(idx & 15);
    int lane = (int)((idx >> 4) & 31);
    int blk  = (int)(idx >> 9);
    int kb   = blk % 10;
    int mt   = (blk / 10) & 3;
    int t    = blk / 40;
    int K    = kb * 32 + a_ko(lane, j);
    int b    = mt * 16 + (lane & 15);
    xpk[idx] = f2bf(x[((size_t)b * NT + t) * ND + K]);
}

// ---------------------------------------------------------------------------
// Pack W = [W_ih | W_hh]^T into bf16 B-fragment layout [64 ntile][KB][32][16]
// ---------------------------------------------------------------------------
__global__ void pack_w_kernel(const float* __restrict__ Wih, const float* __restrict__ Whh,
                              int Dsplit, int KB, ushort_t* __restrict__ out) {
    size_t idx = (size_t)blockIdx.x * blockDim.x + threadIdx.x;
    if (idx >= (size_t)64 * KB * 512) return;
    int j    = (int)(idx & 15);
    int lane = (int)((idx >> 4) & 31);
    int blk  = (int)(idx >> 9);
    int kb   = blk % KB;
    int nt   = blk / KB;
    int N    = nt * 16 + (lane & 15);
    int K    = kb * 32 + a_ko(lane, j);
    float v  = (K < Dsplit) ? Wih[(size_t)N * Dsplit + K]
                            : Whh[(size_t)N * NH + (K - Dsplit)];
    out[idx] = f2bf(v);
}

__global__ void prep_bias_kernel(const float* bih0, const float* bhh0,
                                 const float* bih1, const float* bhh1,
                                 float* bias0, float* bias1) {
    int i = blockIdx.x * blockDim.x + threadIdx.x;
    if (i < 1024) { bias0[i] = bih0[i] + bhh0[i]; bias1[i] = bih1[i] + bhh1[i]; }
}

__global__ void init_state_kernel(ushort_t* hapk, int* cnt) {
    int i = blockIdx.x * blockDim.x + threadIdx.x;
    if (i < 4 * 8 * 512) hapk[i] = 0;
    if (i == 0)          *cnt    = 0;
}

// ---------------------------------------------------------------------------
// Grid-wide barrier: monotone counter; all NWG blocks arrive at index k,
// spin until counter reaches NWG*k. Counter zeroed per scan launch.
// ---------------------------------------------------------------------------
__device__ __forceinline__ void gbar(int* cnt, int k) {
    __threadfence();                 // release h writes (global_wb)
    __syncthreads();
    if (threadIdx.x == 0) {
        atomicAdd(cnt, 1);
        int target = NWG * k;
        while (*(volatile int*)cnt < target) __builtin_amdgcn_s_sleep(1);
    }
    __syncthreads();
    __threadfence();                 // acquire remote h writes (global_inv)
}

// ---------------------------------------------------------------------------
// Persistent LSTM scan. 16 WGs x 16 waves. Wave (gate, m_tile) computes the
// 16x16 tile (rows m_tile*16.., cols gate*256 + q*16..).
// ---------------------------------------------------------------------------
template <int LAYER>
__global__ __launch_bounds__(THR, 1) void lstm_scan_kernel(
        const ushort_t* __restrict__ xpk,   // layer0: packed x; layer1: h1pk
        ushort_t*       __restrict__ h1pk,  // layer0 writes; layer1 unused
        const ushort_t* __restrict__ wpk,
        const float*    __restrict__ bias,
        ushort_t* __restrict__ hapk, float* __restrict__ h2, int* cnt)
{
    constexpr int KBX  = (LAYER == 0) ? 10 : 8;  // x-part k-blocks
    constexpr int KB   = KBX + 8;                // + recurrent (256/32)
    constexpr int SX16 = KBX * 256;              // 16B chunks in one x slice
    constexpr int SREG = SX16 / THR;             // staging chunks per thread

    __shared__ ushort_t xlds[4 * KBX * 512];     // staged x slice (40/32 KB)
    __shared__ float lds_g[4][64][16];           // activated gates, u-slice
    __shared__ float lds_c[64][16];              // persistent cell state slice

    const int lane   = threadIdx.x & 31;
    const int wv     = threadIdx.x >> 5;         // 0..15
    const int m_tile = wv & 3;
    const int gate   = wv >> 2;                  // 0=i 1=f 2=g 3=o
    const int q      = blockIdx.x;               // hidden-slice index 0..15
    const int n_tile = gate * 16 + q;
    const float bcol = bias[n_tile * 16 + (lane & 15)];

    // ---- prologue: zero cell state, stage x slice for t=0 ----
    for (int i = threadIdx.x; i < 64 * 16; i += THR) (&lds_c[0][0])[i] = 0.0f;
    {
        const uint4* src = reinterpret_cast<const uint4*>(xpk);
        uint4*       dst = reinterpret_cast<uint4*>(xlds);
#pragma unroll
        for (int i = 0; i < SREG; ++i)
            dst[threadIdx.x + i * THR] = src[threadIdx.x + i * THR];
    }
    __syncthreads();

    // ---- hoist all weight fragments for this n_tile into VGPRs ----
    v16bf bfr[KB];
#pragma unroll
    for (int kb = 0; kb < KB; ++kb)
        bfr[kb] = *reinterpret_cast<const v16bf*>(
            wpk + ((size_t)(n_tile * KB + kb) * 512) + lane * 16);

    const ushort_t* hap  = hapk + (size_t)m_tile * 8 * 512 + lane * 16;
    const ushort_t* xwav = xlds + (size_t)m_tile * KBX * 512 + lane * 16;

    for (int t = 0; t < NT; ++t) {
        // ---- 1) issue all recurrent h-fragment loads (distinct VGPRs),
        //         pinned at loop top so they fly under the x-part WMMAs ----
        v16bf hfr[8];
#pragma unroll
        for (int i = 0; i < 8; ++i)
            hfr[i] = *reinterpret_cast<const v16bf*>(hap + (size_t)i * 512);
#if __has_builtin(__builtin_amdgcn_sched_barrier)
        __builtin_amdgcn_sched_barrier(0);   // nothing crosses: loads stay up
#endif

        v8f acc0, acc1;
#pragma unroll
        for (int r = 0; r < 8; ++r) { acc0[r] = bcol; acc1[r] = 0.0f; }

        // ---- 2) x-part WMMAs from LDS (hides h-load latency) ----
#pragma unroll
        for (int kb = 0; kb < KBX; ++kb) {
            v16bf a = *reinterpret_cast<const v16bf*>(xwav + (size_t)kb * 512);
            if (kb & 1)
                acc1 = __builtin_amdgcn_wmma_f32_16x16x32_bf16(
                    false, a, false, bfr[kb], (short)0, acc1, false, false);
            else
                acc0 = __builtin_amdgcn_wmma_f32_16x16x32_bf16(
                    false, a, false, bfr[kb], (short)0, acc0, false, false);
        }
        __syncthreads();   // xlds fully consumed; safe to restage later

        // ---- 3) recurrent WMMAs consume prefetched h fragments ----
#pragma unroll
        for (int i = 0; i < 8; ++i) {
            if ((KBX + i) & 1)
                acc1 = __builtin_amdgcn_wmma_f32_16x16x32_bf16(
                    false, hfr[i], false, bfr[KBX + i], (short)0, acc1, false, false);
            else
                acc0 = __builtin_amdgcn_wmma_f32_16x16x32_bf16(
                    false, hfr[i], false, bfr[KBX + i], (short)0, acc0, false, false);
        }

        // ---- L2-warm prefetch two steps ahead (gate-0 waves only) ----
        if (gate == 0 && t + 2 < NT) {
            const ushort_t* xn = xpk + ((size_t)(t + 2) * 4 + m_tile) * KBX * 512 + lane * 16;
#pragma unroll
            for (int kb = 0; kb < KBX; ++kb)
                __builtin_prefetch(xn + (size_t)kb * 512, 0, 0);
        }

        // ---- merge accumulators + activation into LDS gate buffer ----
#pragma unroll
        for (int r = 0; r < 8; ++r) {
            int row = m_tile * 16 + (lane >> 4) * 8 + r;
            float v = acc0[r] + acc1[r];
            float g = (gate == 2) ? tanhf(v) : 1.0f / (1.0f + __expf(-v));
            lds_g[gate][row][lane & 15] = g;
        }
        __syncthreads();

        // ---- 4) issue staging loads for next step's x slice (hfr/acc dead
        //         here -> no spill pressure; L2 latency overlaps cell update)
        uint4 stg[SREG];
        if (t + 1 < NT) {
            const uint4* nsrc = reinterpret_cast<const uint4*>(
                xpk + (size_t)(t + 1) * 4 * KBX * 512);
#pragma unroll
            for (int i = 0; i < SREG; ++i)
                stg[i] = nsrc[threadIdx.x + i * THR];
        }

        // ---- cell update for this WG's slice: 1024 elems / 512 threads ----
#pragma unroll
        for (int rep = 0; rep < 2; ++rep) {
            int e  = threadIdx.x + rep * THR;
            int b  = e >> 4, uq = e & 15;
            int u  = q * 16 + uq;
            float iv = lds_g[0][b][uq], fv = lds_g[1][b][uq];
            float gv = lds_g[2][b][uq], ov = lds_g[3][b][uq];
            float c  = fv * lds_c[b][uq] + iv * gv;
            lds_c[b][uq] = c;
            float h  = ov * tanhf(c);
            ushort_t hb = f2bf(h);
            int mt = b >> 4, m = b & 15, kb = u >> 5, ko = u & 31;
            int half, j;
            if (ko < 16) { half = ko >> 3; j = ko & 7; }
            else         { half = (ko - 16) >> 3; j = 8 + ((ko - 16) & 7); }
            int fo = (mt * 8 + kb) * 512 + (half * 16 + m) * 16 + j;
            hapk[fo] = hb;
            if (LAYER == 0)        h1pk[(size_t)t * 16384 + fo] = hb;
            else if (t == NT - 1)  h2[b * NH + u] = h;
        }

        // ---- commit staged x slice to LDS (consumed after the barrier) ----
        if (t + 1 < NT) {
            uint4* dst = reinterpret_cast<uint4*>(xlds);
#pragma unroll
            for (int i = 0; i < SREG; ++i)
                dst[threadIdx.x + i * THR] = stg[i];
        }

        gbar(cnt, t + 1);   // single grid sync per timestep
    }
}

// ---------------------------------------------------------------------------
// Final FC head: [64,256] x [256,10] + b
// ---------------------------------------------------------------------------
__global__ void fc_kernel(const float* __restrict__ h2, const float* __restrict__ Wfc,
                          const float* __restrict__ bfc, float* __restrict__ out) {
    int tid = blockIdx.x * blockDim.x + threadIdx.x;
    if (tid >= NB * NC) return;
    int b = tid / NC, c = tid % NC;
    float s = bfc[c];
#pragma unroll 8
    for (int u = 0; u < NH; ++u) s += h2[b * NH + u] * Wfc[c * NH + u];
    out[b * NC + c] = s;
}

extern "C" void kernel_launch(void* const* d_in, const int* in_sizes, int n_in,
                              void* d_out, int out_size, void* d_ws, size_t ws_size,
                              hipStream_t stream) {
    (void)in_sizes; (void)n_in; (void)out_size; (void)ws_size;
    const float* x    = (const float*)d_in[0];
    const float* Wih0 = (const float*)d_in[1];
    const float* Whh0 = (const float*)d_in[2];
    const float* bih0 = (const float*)d_in[3];
    const float* bhh0 = (const float*)d_in[4];
    const float* Wih1 = (const float*)d_in[5];
    const float* Whh1 = (const float*)d_in[6];
    const float* bih1 = (const float*)d_in[7];
    const float* bhh1 = (const float*)d_in[8];
    const float* Wfc  = (const float*)d_in[9];
    const float* bfc  = (const float*)d_in[10];

    char* ws = (char*)d_ws;
    ushort_t* xpk   = (ushort_t*)(ws + OFF_XPK);
    ushort_t* h1pk  = (ushort_t*)(ws + OFF_H1PK);
    ushort_t* wpk0  = (ushort_t*)(ws + OFF_WPK0);
    ushort_t* wpk1  = (ushort_t*)(ws + OFF_WPK1);
    float*    bias0 = (float*)(ws + OFF_B0);
    float*    bias1 = (float*)(ws + OFF_B1);
    ushort_t* hapk  = (ushort_t*)(ws + OFF_HAPK);
    float*    h2    = (float*)(ws + OFF_H2);
    int*      cnt   = (int*)(ws + OFF_CNT);

    // pack inputs/weights to bf16 fragment layouts
    pack_x_kernel<<<(2048UL * 4 * 10 * 512 + 255) / 256, 256, 0, stream>>>(x, xpk);
    pack_w_kernel<<<(64UL * 18 * 512 + 255) / 256, 256, 0, stream>>>(Wih0, Whh0, ND, 18, wpk0);
    pack_w_kernel<<<(64UL * 16 * 512 + 255) / 256, 256, 0, stream>>>(Wih1, Whh1, NH, 16, wpk1);
    prep_bias_kernel<<<4, 256, 0, stream>>>(bih0, bhh0, bih1, bhh1, bias0, bias1);

    // layer 0 scan
    init_state_kernel<<<64, 256, 0, stream>>>(hapk, cnt);
    lstm_scan_kernel<0><<<NWG, THR, 0, stream>>>(xpk, h1pk, wpk0, bias0, hapk, h2, cnt);
    // layer 1 scan (consumes h1pk as its packed "x")
    init_state_kernel<<<64, 256, 0, stream>>>(hapk, cnt);
    lstm_scan_kernel<1><<<NWG, THR, 0, stream>>>(h1pk, nullptr, wpk1, bias1, hapk, h2, cnt);
    // classifier head
    fc_kernel<<<3, 256, 0, stream>>>(h2, Wfc, bfc, (float*)d_out);
}
